// DepthMemoryReuseRowQKVAttention_30313879175631
// MI455X (gfx1250) — matrix operand order repaired and checked by reading, not compiled, not run on hardware
//
#include <hip/hip_runtime.h>
#include <hip/hip_bf16.h>

// MI455X / gfx1250, wave32. All GEMM-shaped math on v_wmma_f32_16x16x32_bf16.
// Row-major LDS staging goes through GLOBAL_LOAD_ASYNC_TO_LDS_B128 (ASYNCcnt)
// when the toolchain exposes the builtin; falls back to load+ds_store.

typedef __bf16 bf16;
typedef __attribute__((ext_vector_type(16))) __bf16 v16bf;
typedef __attribute__((ext_vector_type(8)))  float  v8f;

#define BATCH   2
#define SEQ     2048
#define DMODEL  1024
#define NHEADS  16
#define HDIM    64
#define LMEM    8

#if __has_builtin(__builtin_amdgcn_global_load_async_to_lds_b128)
#define HAVE_ASYNC_LDS 1
#else
#define HAVE_ASYNC_LDS 0
#endif

// builtin's param type per hipcc diagnostic: 'int __attribute__((vector_size(16))) __device__ *'
typedef int v4i_as __attribute__((vector_size(4 * sizeof(int))));
typedef __attribute__((address_space(1))) v4i_as* g_v4i_ptr;
typedef __attribute__((address_space(3))) v4i_as* l_v4i_ptr;

__device__ __forceinline__ void async_copy_b128(const void* gp, void* lp) {
#if HAVE_ASYNC_LDS
  __builtin_amdgcn_global_load_async_to_lds_b128((g_v4i_ptr)gp, (l_v4i_ptr)lp, 0, 0);
#else
  *(uint4*)lp = *(const uint4*)gp;
#endif
}

__device__ __forceinline__ void async_copy_wait() {
#if HAVE_ASYNC_LDS
#if __has_builtin(__builtin_amdgcn_s_wait_asynccnt)
  __builtin_amdgcn_s_wait_asynccnt(0);
#else
  asm volatile("s_wait_asynccnt 0" ::: "memory");
#endif
#endif
}

union Frag16 { v16bf v; uint4 u[2]; bf16 h[16]; };
union BF16x8 { uint4 u; bf16 h[8]; };
union BF16x4 { uint2 u; bf16 h[4]; };

__device__ __forceinline__ v8f wmma_bf16f32(v16bf a, v16bf b, v8f c) {
  // 8 args: (neg_a, A, neg_b, B, c_mod, C, reuse_a, reuse_b)
  return __builtin_amdgcn_wmma_f32_16x16x32_bf16(false, a, false, b, (short)0, c, false, false);
}

// ---------------------------------------------------------------- convert
__global__ __launch_bounds__(256) void cvt4_kernel(const float* __restrict__ s,
                                                   bf16* __restrict__ d, int n4) {
  int i = blockIdx.x * 256 + threadIdx.x;
  if (i < n4) {
    float4 f = ((const float4*)s)[i];
    BF16x4 o;
    o.h[0] = (bf16)f.x; o.h[1] = (bf16)f.y; o.h[2] = (bf16)f.z; o.h[3] = (bf16)f.w;
    ((uint2*)d)[i] = o.u;
  }
}

// -------------------------------------------------- shared WMMA GEMM core
// Block: 256 threads = 8 waves (4 in M x 2 in N). Block tile 64(M) x 128(N).
// Per wave: 16(M) x 64(N) via 4 f32 accumulators, K stepped by 32.
template <int LDB_N>
__device__ __forceinline__ void gemm_tile_256(const bf16* __restrict__ Abf,
                                              const bf16* __restrict__ Bbf,
                                              int K, int bm, int bn,
                                              bf16 (&As)[64][32], bf16 (&Bs)[128][40],
                                              v8f (&acc)[4]) {
  const int tid  = threadIdx.x;
  const int lane = tid & 31, wave = tid >> 5;
  const int waveM = wave & 3, waveN = wave >> 2;
  const int hl = lane >> 4, l15 = lane & 15;
  const int ar = tid >> 2, ac = (tid & 3) * 8;    // A stage: 8 bf16 / thread
  const int bk = tid >> 3, bc = (tid & 7) * 16;   // B stage: 16 bf16 / thread

  for (int kk = 0; kk < K; kk += 32) {
    __syncthreads();
    // A tile: straight row-major copy -> async DMA into LDS
    async_copy_b128(&Abf[(size_t)(bm + ar) * K + kk + ac], &As[ar][ac]);
    // B tile: needs transpose -> manual staging
    BF16x8 t0, t1;
    const bf16* bp = &Bbf[(size_t)(kk + bk) * LDB_N + bn + bc];
    t0.u = *(const uint4*)bp;
    t1.u = *(const uint4*)(bp + 8);
#pragma unroll
    for (int i = 0; i < 8; ++i) { Bs[bc + i][bk] = t0.h[i]; Bs[bc + 8 + i][bk] = t1.h[i]; }
    if (kk + 32 < K) {
      __builtin_prefetch(&Abf[(size_t)(bm + ar) * K + kk + 32 + ac], 0, 1);
      __builtin_prefetch(&Bbf[(size_t)(kk + 32 + bk) * LDB_N + bn + bc], 0, 1);
    }
    async_copy_wait();
    __syncthreads();

    Frag16 a;  // A 16x32 bf16 layout: lanes<16 hold K{0-7,16-23}, lanes>=16 K{8-15,24-31}
    a.u[0] = *(const uint4*)&As[waveM * 16 + l15][hl * 8];
    a.u[1] = *(const uint4*)&As[waveM * 16 + l15][16 + hl * 8];
#pragma unroll
    for (int g = 0; g < 4; ++g) {
      Frag16 b;  // B 32x16 layout: lanes<16 hold K{0-7,16-23}, lanes>=16 K{8-15,24-31} (col = lane&15)
      const bf16* bsp = &Bs[waveN * 64 + g * 16 + l15][hl * 8];
      b.u[0] = *(const uint4*)bsp;
      b.u[1] = *(const uint4*)(bsp + 16);
      acc[g] = wmma_bf16f32(a.v, b.v, acc[g]);
    }
  }
}

// ---------------------------------------------------------- QKV projection
// qkv = x @ W_qkv + b_qkv ; split into q(bf16 scratch), k/v (fp32 outputs + bf16 scratch)
__global__ __launch_bounds__(256)
void qkv_gemm_kernel(const bf16* __restrict__ Abf, const bf16* __restrict__ Bbf,
                     const float* __restrict__ bias,
                     float* __restrict__ kout, float* __restrict__ vout,
                     bf16* __restrict__ qbf, bf16* __restrict__ kbf, bf16* __restrict__ vbf) {
  __shared__ __attribute__((aligned(16))) bf16 As[64][32];
  __shared__ __attribute__((aligned(16))) bf16 Bs[128][40];
  const int bm = blockIdx.y * 64, bn = blockIdx.x * 128;
  v8f acc[4] = {};
  gemm_tile_256<3 * DMODEL>(Abf, Bbf, DMODEL, bm, bn, As, Bs, acc);

  const int lane = threadIdx.x & 31, wave = threadIdx.x >> 5;
  const int waveM = wave & 3, waveN = wave >> 2;
  const int hl = lane >> 4, l15 = lane & 15;
#pragma unroll
  for (int g = 0; g < 4; ++g) {
    const int c = bn + waveN * 64 + g * 16 + l15;
    const float bv = bias[c];
#pragma unroll
    for (int j = 0; j < 8; ++j) {
      const int m = bm + waveM * 16 + hl * 8 + j;     // row = b*SEQ + s
      const float val = acc[g][j] + bv;
      const int bb = m >> 11, ss = m & (SEQ - 1);
      if (c < DMODEL) {
        const int h = c >> 6, d = c & 63;
        qbf[(((size_t)(bb * NHEADS + h) * SEQ) + ss) * HDIM + d] = (bf16)val;
      } else if (c < 2 * DMODEL) {
        const int c2 = c - DMODEL, h = c2 >> 6, d = c2 & 63;
        const size_t off = (((size_t)(bb * NHEADS + h) * SEQ) + ss) * HDIM + d;
        kout[off] = val; kbf[off] = (bf16)val;
      } else {
        const int c2 = c - 2 * DMODEL, h = c2 >> 6, d = c2 & 63;
        const size_t off = (((size_t)(bb * NHEADS + h) * SEQ) + ss) * HDIM + d;
        vout[off] = val; vbf[off] = (bf16)val;
      }
    }
  }
}

// ---------------------------------------------------------------- attention
// Block: 128 threads = 4 waves, each owns 16 query rows of a 64-row q block.
// Online softmax over [causal token scores | 8 memory-slot scores].
__global__ __launch_bounds__(128)
void attn_kernel(const bf16* __restrict__ qbf, const bf16* __restrict__ kbf,
                 const bf16* __restrict__ vbf,
                 const float* __restrict__ past_keys, const float* __restrict__ past_values,
                 bf16* __restrict__ ctx_bf) {
  __shared__ __attribute__((aligned(16))) bf16 Ks[32][64];   // (key, d)
  __shared__ __attribute__((aligned(16))) bf16 Vt[64][40];   // (d, key) transposed, padded
  __shared__ __attribute__((aligned(16))) bf16 Ps[4][16][40];// per-wave P tile, padded
  __shared__ float Mw[4][16][LMEM];
  __shared__ float Mm[4][16];
  __shared__ float Ml[4][16];

  const float SC = 0.125f * 1.44269504088896f;  // (1/sqrt(HDIM)) * log2(e)
  const int tid = threadIdx.x;
  const int lane = tid & 31, wave = tid >> 5;
  const int hl = lane >> 4, l15 = lane & 15;
  const int bh = blockIdx.z * NHEADS + blockIdx.y;
  const int q0 = blockIdx.x * 64;
  const int qt = q0 + wave * 16;

  // ---- Q fragments (d = 0..31 and 32..63), per documented A-matrix layout
  Frag16 aQ[2];
  {
    const bf16* qp = &qbf[((size_t)bh * SEQ + qt + l15) * HDIM];
#pragma unroll
    for (int dc = 0; dc < 2; ++dc) {
      aQ[dc].u[0] = *(const uint4*)(qp + dc * 32 + hl * 8);
      aQ[dc].u[1] = *(const uint4*)(qp + dc * 32 + 16 + hl * 8);
    }
  }

  // ---- memory-slot scores: lane pair (2 lanes) per query row
  {
    const int r2 = lane >> 1, ph = lane & 1;
    float qv[32];
    const bf16* qrow = &qbf[((size_t)bh * SEQ + qt + r2) * HDIM + ph * 32];
#pragma unroll
    for (int i = 0; i < 32; ++i) qv[i] = (float)qrow[i];
    float ms[LMEM];
#pragma unroll
    for (int l = 0; l < LMEM; ++l) {
      const float* pk = &past_keys[(((size_t)bh * SEQ + qt + r2) * LMEM + l) * HDIM + ph * 32];
      float dsum = 0.f;
#pragma unroll
      for (int i = 0; i < 32; ++i) dsum += qv[i] * pk[i];
      dsum += __shfl_xor(dsum, 1, 32);
      ms[l] = dsum * SC;
    }
    float m0 = ms[0];
#pragma unroll
    for (int l = 1; l < LMEM; ++l) m0 = fmaxf(m0, ms[l]);
    float l0 = 0.f;
#pragma unroll
    for (int l = 0; l < LMEM; ++l) {
      const float w = exp2f(ms[l] - m0);
      l0 += w;
      if (ph == 0) Mw[wave][r2][l] = w;
    }
    if (ph == 0) { Mm[wave][r2] = m0; Ml[wave][r2] = l0; }
  }
  __syncthreads();

  // ---- init state + memory-slot context in WMMA C layout
  float mstate[8], lstate[8];
  v8f acc[4] = {};
#pragma unroll
  for (int j = 0; j < 8; ++j) {
    const int row = hl * 8 + j;
    mstate[j] = Mm[wave][row];
    lstate[j] = Ml[wave][row];
    float wl[LMEM];
#pragma unroll
    for (int l = 0; l < LMEM; ++l) wl[l] = Mw[wave][row][l];
    const float* pvb = &past_values[(((size_t)bh * SEQ + qt + row) * LMEM) * HDIM];
#pragma unroll
    for (int g = 0; g < 4; ++g) {
      const int col = g * 16 + l15;
      float a = 0.f;
#pragma unroll
      for (int l = 0; l < LMEM; ++l) a += wl[l] * pvb[l * HDIM + col];
      acc[g][j] = a;
    }
  }

  // ---- stream token keys/values, 32 per iteration
  const int kend = q0 + 64;
  const int kr = tid >> 2, dc = (tid & 3) * 16;
  for (int k0 = 0; k0 < kend; k0 += 32) {
    __syncthreads();
    {  // cooperative staging: K row-major (async DMA), V transposed (manual)
      const bf16* kp = &kbf[((size_t)bh * SEQ + k0 + kr) * HDIM + dc];
      const bf16* vp = &vbf[((size_t)bh * SEQ + k0 + kr) * HDIM + dc];
      async_copy_b128(kp,     &Ks[kr][dc]);
      async_copy_b128(kp + 8, &Ks[kr][dc + 8]);
      BF16x8 v0, v1;
      v0.u = *(const uint4*)vp;
      v1.u = *(const uint4*)(vp + 8);
#pragma unroll
      for (int i = 0; i < 8; ++i) { Vt[dc + i][kr] = v0.h[i]; Vt[dc + 8 + i][kr] = v1.h[i]; }
      async_copy_wait();
    }
    __syncthreads();

    // scores for two 16-key subtiles (K over d split 2x32)
    float sr[2][8];
#pragma unroll
    for (int t = 0; t < 2; ++t) {
      v8f s = {};
#pragma unroll
      for (int dcc = 0; dcc < 2; ++dcc) {
        Frag16 bK;
        const bf16* kp = &Ks[t * 16 + l15][dcc * 32 + hl * 8];
        bK.u[0] = *(const uint4*)kp;
        bK.u[1] = *(const uint4*)(kp + 16);
        s = wmma_bf16f32(aQ[dcc].v, bK.v, s);
      }
      const int kcol = k0 + t * 16 + l15;
#pragma unroll
      for (int j = 0; j < 8; ++j) {
        const int row = qt + hl * 8 + j;
        sr[t][j] = (kcol <= row) ? s[j] * SC : -3.0e38f;
      }
    }

    // online softmax update (row reductions across 16-lane halves)
    float alpha[8];
#pragma unroll
    for (int j = 0; j < 8; ++j) {
      float tm = fmaxf(sr[0][j], sr[1][j]);
#pragma unroll
      for (int off = 8; off >= 1; off >>= 1) tm = fmaxf(tm, __shfl_xor(tm, off, 32));
      const float mn = fmaxf(mstate[j], tm);
      alpha[j] = exp2f(mstate[j] - mn);
      const float p0 = exp2f(sr[0][j] - mn);
      const float p1 = exp2f(sr[1][j] - mn);
      sr[0][j] = p0; sr[1][j] = p1;
      float r = p0 + p1;
#pragma unroll
      for (int off = 8; off >= 1; off >>= 1) r += __shfl_xor(r, off, 32);
      lstate[j] = lstate[j] * alpha[j] + r;
      mstate[j] = mn;
    }
#pragma unroll
    for (int g = 0; g < 4; ++g)
#pragma unroll
      for (int j = 0; j < 8; ++j) acc[g][j] *= alpha[j];

    // P tile -> LDS (C layout -> A layout repack), then P @ V
#pragma unroll
    for (int j = 0; j < 8; ++j) {
      const int row = hl * 8 + j;
      Ps[wave][row][l15]      = (bf16)sr[0][j];
      Ps[wave][row][16 + l15] = (bf16)sr[1][j];
    }
    Frag16 aP;
    aP.u[0] = *(const uint4*)&Ps[wave][l15][hl * 8];
    aP.u[1] = *(const uint4*)&Ps[wave][l15][16 + hl * 8];
#pragma unroll
    for (int g = 0; g < 4; ++g) {
      Frag16 bV;
      const bf16* vp = &Vt[g * 16 + l15][hl * 8];
      bV.u[0] = *(const uint4*)vp;
      bV.u[1] = *(const uint4*)(vp + 16);
      acc[g] = wmma_bf16f32(aP.v, bV.v, acc[g]);
    }
  }

  // ---- normalize, write merged-head context (bf16) for output GEMM
  const int bb = bh >> 4, h = bh & 15;
#pragma unroll
  for (int g = 0; g < 4; ++g)
#pragma unroll
    for (int j = 0; j < 8; ++j) {
      const int row = qt + hl * 8 + j;
      const float val = acc[g][j] / lstate[j];
      ctx_bf[((size_t)(bb * SEQ + row)) * DMODEL + h * HDIM + g * 16 + l15] = (bf16)val;
    }
}

// ----------------------------------------------------------- out projection
__global__ __launch_bounds__(256)
void outproj_gemm_kernel(const bf16* __restrict__ Abf, const bf16* __restrict__ Bbf,
                         const float* __restrict__ bias, float* __restrict__ out) {
  __shared__ __attribute__((aligned(16))) bf16 As[64][32];
  __shared__ __attribute__((aligned(16))) bf16 Bs[128][40];
  const int bm = blockIdx.y * 64, bn = blockIdx.x * 128;
  v8f acc[4] = {};
  gemm_tile_256<DMODEL>(Abf, Bbf, DMODEL, bm, bn, As, Bs, acc);

  const int lane = threadIdx.x & 31, wave = threadIdx.x >> 5;
  const int waveM = wave & 3, waveN = wave >> 2;
  const int hl = lane >> 4, l15 = lane & 15;
#pragma unroll
  for (int g = 0; g < 4; ++g) {
    const int c = bn + waveN * 64 + g * 16 + l15;
    const float bv = bias[c];
#pragma unroll
    for (int j = 0; j < 8; ++j) {
      const int m = bm + waveM * 16 + hl * 8 + j;
      out[(size_t)m * DMODEL + c] = acc[g][j] + bv;
    }
  }
}

// ------------------------------------------------------------------- launch
extern "C" void kernel_launch(void* const* d_in, const int* in_sizes, int n_in,
                              void* d_out, int out_size, void* d_ws, size_t ws_size,
                              hipStream_t stream) {
  (void)in_sizes; (void)n_in; (void)out_size; (void)ws_size;
  const float* x           = (const float*)d_in[0];
  const float* past_keys   = (const float*)d_in[1];
  const float* past_values = (const float*)d_in[2];
  const float* W_qkv       = (const float*)d_in[3];
  const float* b_qkv       = (const float*)d_in[4];
  const float* W_out       = (const float*)d_in[5];
  const float* b_out       = (const float*)d_in[6];

  float* out  = (float*)d_out;                         // (2,2048,1024)
  float* kout = out + (size_t)BATCH * SEQ * DMODEL;    // (2,16,2048,64)
  float* vout = kout + (size_t)BATCH * SEQ * DMODEL;   // (2,16,2048,64)

  char* ws = (char*)d_ws;
  bf16* x_bf    = (bf16*)ws; ws += (size_t)BATCH * SEQ * DMODEL * 2;
  bf16* wqkv_bf = (bf16*)ws; ws += (size_t)DMODEL * 3 * DMODEL * 2;
  bf16* wout_bf = (bf16*)ws; ws += (size_t)DMODEL * DMODEL * 2;
  bf16* q_bf    = (bf16*)ws; ws += (size_t)BATCH * SEQ * DMODEL * 2;
  bf16* k_bf    = (bf16*)ws; ws += (size_t)BATCH * SEQ * DMODEL * 2;
  bf16* v_bf    = (bf16*)ws; ws += (size_t)BATCH * SEQ * DMODEL * 2;
  bf16* ctx_bf  = (bf16*)ws; ws += (size_t)BATCH * SEQ * DMODEL * 2;

  const int n4x = BATCH * SEQ * DMODEL / 4;
  const int n4q = DMODEL * 3 * DMODEL / 4;
  const int n4o = DMODEL * DMODEL / 4;
  cvt4_kernel<<<(n4x + 255) / 256, 256, 0, stream>>>(x, x_bf, n4x);
  cvt4_kernel<<<(n4q + 255) / 256, 256, 0, stream>>>(W_qkv, wqkv_bf, n4q);
  cvt4_kernel<<<(n4o + 255) / 256, 256, 0, stream>>>(W_out, wout_bf, n4o);

  qkv_gemm_kernel<<<dim3(3 * DMODEL / 128, BATCH * SEQ / 64), 256, 0, stream>>>(
      x_bf, wqkv_bf, b_qkv, kout, vout, q_bf, k_bf, v_bf);

  attn_kernel<<<dim3(SEQ / 64, NHEADS, BATCH), 128, 0, stream>>>(
      q_bf, k_bf, v_bf, past_keys, past_values, ctx_bf);

  outproj_gemm_kernel<<<dim3(DMODEL / 128, BATCH * SEQ / 64), 256, 0, stream>>>(
      ctx_bf, wout_bf, b_out, out);
}
